// backbone_pointnet2_8083128451498
// MI455X (gfx1250) — matrix operand order repaired and checked by reading, not compiled, not run on hardware
//
#include <hip/hip_runtime.h>

#define CDIV(a,b) (((a)+(b)-1)/(b))

// ---------------------------------------------------------------------------
// WMMA types (CDNA5 / gfx1250, wave32)
// ---------------------------------------------------------------------------
typedef __attribute__((ext_vector_type(16))) __bf16          v16bf;
typedef __attribute__((ext_vector_type(16))) unsigned short  v16us;
typedef __attribute__((ext_vector_type(8)))  unsigned short  v8us;
typedef __attribute__((ext_vector_type(8)))  float           v8f;

__device__ __forceinline__ unsigned short f32_to_bf16(float f) {
  unsigned u = __builtin_bit_cast(unsigned, f);
  u += 0x7FFFu + ((u >> 16) & 1u);          // round-to-nearest-even
  return (unsigned short)(u >> 16);
}

// ---------------------------------------------------------------------------
// Pack W[K,N] (f32) into WMMA B-fragment order, bf16, zero padded:
//   Wp[((nt*KT + kt)*32 + lane)*16 + e] = bf16(W[(kt*32 + (e>>3)*16 + (lane>>4)*8 + (e&7))*N
//                                               + nt*16 + (lane&15)])
// so each GEMM lane reads one contiguous 32B chunk per K-tile.
// ---------------------------------------------------------------------------
__global__ void k_pack_w(const float* __restrict__ W, unsigned short* __restrict__ Wp,
                         int K, int N, int KT, int NT)
{
  size_t total = (size_t)NT * KT * 512;
  size_t i = (size_t)blockIdx.x * blockDim.x + threadIdx.x;
  if (i >= total) return;
  int e    = (int)(i & 15);
  int lane = (int)((i >> 4) & 31);
  size_t chunk = i >> 9;
  int kt = (int)(chunk % KT);
  int nt = (int)(chunk / KT);
  int k = kt * 32 + (e >> 3) * 16 + (lane >> 4) * 8 + (e & 7);
  int n = nt * 16 + (lane & 15);
  Wp[i] = (k < K && n < N) ? f32_to_bf16(W[(size_t)k * N + n]) : (unsigned short)0;
}

// Pack X[M,K] (f32) -> Xp[M,Kp] (bf16, zero padded), row-major.
__global__ void k_pack_x(const float* __restrict__ X, unsigned short* __restrict__ Xp,
                         size_t M, int K, int Kp)
{
  size_t total = M * Kp;
  size_t i = (size_t)blockIdx.x * blockDim.x + threadIdx.x;
  if (i >= total) return;
  int k = (int)(i % Kp); size_t r = i / Kp;
  Xp[i] = (k < K) ? f32_to_bf16(X[r * K + k]) : (unsigned short)0;
}

// ---------------------------------------------------------------------------
// Fused GEMM: Y[M,N] = relu( (X @ W) * g[N] + b[N] ), bf16 in / f32 accum.
// 2x2 16x16 tiles per wave (32x32), 4 waves per block (64x64 per block).
// All loads unconditional (clamped addresses); stores masked. EXEC stays
// all-ones across every v_wmma (ISA 7.12 requirement).
// ---------------------------------------------------------------------------
__device__ __forceinline__ void store_tile(const v8f& acc, int m0, int n0, int lane,
                                           const float* __restrict__ g,
                                           const float* __restrict__ bias,
                                           float* __restrict__ Y, int M, int N)
{
  int h = lane >> 4, l = lane & 15;
  int col = n0 + l;
  if (col < N) {
    float gg = g[col], bb = bias[col];
#pragma unroll
    for (int r = 0; r < 8; ++r) {
      int row = m0 + r + 8 * h;
      if (row < M) Y[(size_t)row * N + col] = fmaxf(acc[r] * gg + bb, 0.0f);
    }
  }
}

__global__ void __launch_bounds__(128)
k_gemm_wmma(const unsigned short* __restrict__ Xp, const unsigned short* __restrict__ Wp,
            const float* __restrict__ g, const float* __restrict__ bias,
            float* __restrict__ Y, int M, int N, int Kp, int KT, int NT)
{
  const int lane = threadIdx.x & 31;
  const int wave = threadIdx.x >> 5;         // 2x2 wave grid inside block
  const int wm = wave & 1, wn = wave >> 1;
  const int m0 = blockIdx.x * 64 + wm * 32;  // two 16-row tiles: m0, m0+16
  const int n0 = blockIdx.y * 64 + wn * 32;  // two 16-col tiles: n0, n0+16
  const int h = lane >> 4, l = lane & 15;

  int rA0 = m0 + l;      if (rA0 >= M) rA0 = M - 1;
  int rA1 = m0 + 16 + l; if (rA1 >= M) rA1 = M - 1;
  const unsigned short* a0p = Xp + (size_t)rA0 * Kp + h * 8;
  const unsigned short* a1p = Xp + (size_t)rA1 * Kp + h * 8;

  int nt0 = n0 >> 4;     if (nt0 >= NT) nt0 = NT - 1;
  int nt1 = nt0 + 1;     if (nt1 >= NT) nt1 = NT - 1;
  const unsigned short* b0p = Wp + ((size_t)nt0 * KT * 32 + lane) * 16;
  const unsigned short* b1p = Wp + ((size_t)nt1 * KT * 32 + lane) * 16;

  v8f acc00 = {}, acc01 = {}, acc10 = {}, acc11 = {};
  for (int kt = 0; kt < KT; ++kt) {
    v8us a0lo = *(const v8us*)(a0p);
    v8us a0hi = *(const v8us*)(a0p + 16);
    v8us a1lo = *(const v8us*)(a1p);
    v8us a1hi = *(const v8us*)(a1p + 16);
    v16us b0u = *(const v16us*)(b0p);
    v16us b1u = *(const v16us*)(b1p);
    __builtin_prefetch((const void*)(a0p + 32), 0, 1);
    __builtin_prefetch((const void*)(b0p + 512), 0, 1);
    v16us a0u = __builtin_shufflevector(a0lo, a0hi, 0,1,2,3,4,5,6,7,8,9,10,11,12,13,14,15);
    v16us a1u = __builtin_shufflevector(a1lo, a1hi, 0,1,2,3,4,5,6,7,8,9,10,11,12,13,14,15);
    v16bf a0 = __builtin_bit_cast(v16bf, a0u);
    v16bf a1 = __builtin_bit_cast(v16bf, a1u);
    v16bf b0 = __builtin_bit_cast(v16bf, b0u);
    v16bf b1 = __builtin_bit_cast(v16bf, b1u);
    acc00 = __builtin_amdgcn_wmma_f32_16x16x32_bf16(false, a0, false, b0, (short)0, acc00, false, false);
    acc01 = __builtin_amdgcn_wmma_f32_16x16x32_bf16(false, a0, false, b1, (short)0, acc01, false, false);
    acc10 = __builtin_amdgcn_wmma_f32_16x16x32_bf16(false, a1, false, b0, (short)0, acc10, false, false);
    acc11 = __builtin_amdgcn_wmma_f32_16x16x32_bf16(false, a1, false, b1, (short)0, acc11, false, false);
    a0p += 32; a1p += 32; b0p += 512; b1p += 512;
  }

  store_tile(acc00, m0,      n0,      lane, g, bias, Y, M, N);
  store_tile(acc01, m0,      n0 + 16, lane, g, bias, Y, M, N);
  store_tile(acc10, m0 + 16, n0,      lane, g, bias, Y, M, N);
  store_tile(acc11, m0 + 16, n0 + 16, lane, g, bias, Y, M, N);
}

// ---------------------------------------------------------------------------
// Farthest point sampling. One 1024-thread block per batch; coords + running
// min-dists in registers (<=16 pts/thread); LDS tree argmax (ties: low index).
// ---------------------------------------------------------------------------
#define FPS_T 1024
__global__ void __launch_bounds__(FPS_T)
k_fps(const float* __restrict__ xyz, int* __restrict__ fi, int N, int npoint)
{
  const int b = blockIdx.x;
  const int tid = threadIdx.x;
  const float* P = xyz + (size_t)b * N * 3;
  const int cnt = CDIV(N, FPS_T);

  float px[16], py[16], pz[16], fd[16];
#pragma unroll
  for (int t = 0; t < 16; ++t) {
    int n = tid + t * FPS_T;
    bool ok = (t < cnt) && (n < N);
    px[t] = ok ? P[n * 3 + 0] : 0.0f;
    py[t] = ok ? P[n * 3 + 1] : 0.0f;
    pz[t] = ok ? P[n * 3 + 2] : 0.0f;
    fd[t] = ok ? 1e10f : -1e30f;
  }

  __shared__ float sval[FPS_T];
  __shared__ int   sidx[FPS_T];

  int far = 0;
  for (int it = 0; it < npoint; ++it) {
    if (tid == 0) fi[b * npoint + it] = far;
    float fx = P[far * 3 + 0], fy = P[far * 3 + 1], fz = P[far * 3 + 2];
    float best = -1e30f; int bi = 0x7FFFFFFF;
#pragma unroll
    for (int t = 0; t < 16; ++t) {
      if (t < cnt) {
        int n = tid + t * FPS_T;
        float dx = px[t] - fx, dy = py[t] - fy, dz = pz[t] - fz;
        float d = dx * dx + dy * dy + dz * dz;
        float nd = fminf(fd[t], d);
        fd[t] = nd;
        if (n < N && (nd > best || (nd == best && n < bi))) { best = nd; bi = n; }
      }
    }
    sval[tid] = best; sidx[tid] = bi;
    __syncthreads();
    for (int s = FPS_T / 2; s > 0; s >>= 1) {
      if (tid < s) {
        float ov = sval[tid + s]; int oi = sidx[tid + s];
        if (ov > sval[tid] || (ov == sval[tid] && oi < sidx[tid])) {
          sval[tid] = ov; sidx[tid] = oi;
        }
      }
      __syncthreads();
    }
    far = sidx[0];
    __syncthreads();
  }
}

// ---------------------------------------------------------------------------
// Small data-movement / pooling kernels
// ---------------------------------------------------------------------------
__global__ void k_transpose_points(const float* __restrict__ pts,
                                   float* __restrict__ out, int B, int C, int N)
{ // out[b][n][c] = pts[b][c][n]
  size_t total = (size_t)B * N * C;
  size_t i = (size_t)blockIdx.x * blockDim.x + threadIdx.x;
  if (i >= total) return;
  int c = (int)(i % C); size_t t = i / C;
  int n = (int)(t % N); int b = (int)(t / N);
  out[i] = pts[((size_t)b * C + c) * N + n];
}

__global__ void k_concat2(const float* __restrict__ a, int Ca,
                          const float* __restrict__ b, int Cb,
                          float* __restrict__ out, size_t rows)
{
  int Ct = Ca + Cb;
  size_t total = rows * Ct;
  size_t i = (size_t)blockIdx.x * blockDim.x + threadIdx.x;
  if (i >= total) return;
  int c = (int)(i % Ct); size_t r = i / Ct;
  out[i] = (c < Ca) ? a[r * Ca + c] : b[r * Cb + (c - Ca)];
}

__global__ void k_gather_xyz(const float* __restrict__ xyz, const int* __restrict__ fi,
                             float* __restrict__ nxyz, int B, int N, int S)
{
  int i = blockIdx.x * blockDim.x + threadIdx.x;
  if (i >= B * S) return;
  int b = i / S;
  int n = fi[i];
  const float* src = xyz + ((size_t)b * N + n) * 3;
  float* dst = nxyz + (size_t)i * 3;
  dst[0] = src[0]; dst[1] = src[1]; dst[2] = src[2];
}

__global__ void k_ball_query(const float* __restrict__ nxyz, const float* __restrict__ xyz,
                             int* __restrict__ gi, int B, int S, int N, float r2, int ns)
{
  int i = blockIdx.x * blockDim.x + threadIdx.x;
  if (i >= B * S) return;
  int b = i / S;
  const float* q = nxyz + (size_t)i * 3;
  float qx = q[0], qy = q[1], qz = q[2];
  const float* P = xyz + (size_t)b * N * 3;
  int* out = gi + (size_t)i * ns;
  int cnt = 0, first = -1;
  for (int n = 0; n < N && cnt < ns; ++n) {
    float dx = P[n*3]-qx, dy = P[n*3+1]-qy, dz = P[n*3+2]-qz;
    if (dx*dx + dy*dy + dz*dz <= r2) {
      if (first < 0) first = n;
      out[cnt++] = n;
    }
  }
  if (first < 0) first = N - 1;
  for (; cnt < ns; ++cnt) out[cnt] = first;
}

__global__ void k_group(const float* __restrict__ feats, const int* __restrict__ gi,
                        float* __restrict__ out, int B, int S, int ns, int N, int C)
{ // out[b][s][k][c] = feats[b][gi[b][s][k]][c]
  size_t total = (size_t)B * S * ns * C;
  size_t i = (size_t)blockIdx.x * blockDim.x + threadIdx.x;
  if (i >= total) return;
  int c = (int)(i % C); size_t t = i / C;
  int k = (int)(t % ns); size_t t2 = t / ns;
  int s = (int)(t2 % S); int b = (int)(t2 / S);
  int n = gi[((size_t)b * S + s) * ns + k];
  out[i] = feats[((size_t)b * N + n) * C + c];
}

__global__ void k_maxpool(const float* __restrict__ x, float* __restrict__ y,
                          size_t rows, int ns, int C)
{ // y[r][c] = max_k x[r][k][c]
  size_t total = rows * C;
  size_t i = (size_t)blockIdx.x * blockDim.x + threadIdx.x;
  if (i >= total) return;
  int c = (int)(i % C); size_t r = i / C;
  const float* src = x + (r * (size_t)ns) * C + c;
  float m = -1e30f;
  for (int k = 0; k < ns; ++k) m = fmaxf(m, src[(size_t)k * C]);
  y[i] = m;
}

__global__ void k_tileconcat(const float* __restrict__ gf, const float* __restrict__ f,
                             float* __restrict__ out, int B, int n, int Cg, int Cf)
{ // out[b][i] = [ gf[b], f[b][i] ]
  int Ct = Cg + Cf;
  size_t total = (size_t)B * n * Ct;
  size_t i = (size_t)blockIdx.x * blockDim.x + threadIdx.x;
  if (i >= total) return;
  int c = (int)(i % Ct); size_t r = i / Ct;
  int b = (int)(r / n);
  out[i] = (c < Cg) ? gf[(size_t)b * Cg + c] : f[r * Cf + (c - Cg)];
}

__global__ void k_three_nn(const float* __restrict__ xyz1, const float* __restrict__ xyz2,
                           int* __restrict__ idx3, float* __restrict__ w3,
                           int B, int n1, int n2)
{
  int i = blockIdx.x * blockDim.x + threadIdx.x;
  if (i >= B * n1) return;
  int b = i / n1;
  const float* q = xyz1 + (size_t)i * 3;
  float qx = q[0], qy = q[1], qz = q[2];
  const float* P = xyz2 + (size_t)b * n2 * 3;
  float d0 = 1e30f, d1 = 1e30f, d2 = 1e30f; int i0 = 0, i1 = 0, i2 = 0;
  for (int n = 0; n < n2; ++n) {
    float dx = P[n*3]-qx, dy = P[n*3+1]-qy, dz = P[n*3+2]-qz;
    float d = dx*dx + dy*dy + dz*dz;
    if (d < d0)      { d2=d1;i2=i1; d1=d0;i1=i0; d0=d;i0=n; }
    else if (d < d1) { d2=d1;i2=i1; d1=d;i1=n; }
    else if (d < d2) { d2=d;i2=n; }
  }
  float w0 = 1.f/(d0+1e-8f), w1 = 1.f/(d1+1e-8f), w2 = 1.f/(d2+1e-8f);
  float ws = w0 + w1 + w2;
  w3[(size_t)i*3+0] = w0/ws; w3[(size_t)i*3+1] = w1/ws; w3[(size_t)i*3+2] = w2/ws;
  idx3[(size_t)i*3+0] = i0;  idx3[(size_t)i*3+1] = i1;  idx3[(size_t)i*3+2] = i2;
}

__global__ void k_interp_concat(const float* __restrict__ f2, const int* __restrict__ idx3,
                                const float* __restrict__ w3, const float* __restrict__ skip,
                                float* __restrict__ out, int B, int n1, int n2, int Ci, int Cs)
{ // out[b][p] = [ sum_k w*f2[b][idx_k], skip[b][p] ]
  int Ct = Ci + Cs;
  size_t total = (size_t)B * n1 * Ct;
  size_t i = (size_t)blockIdx.x * blockDim.x + threadIdx.x;
  if (i >= total) return;
  int c = (int)(i % Ct); size_t r = i / Ct;
  int b = (int)(r / n1);
  float v;
  if (c < Ci) {
    const int*   id = idx3 + r * 3;
    const float* w  = w3   + r * 3;
    size_t base = (size_t)b * n2;
    v = w[0] * f2[(base + id[0]) * Ci + c]
      + w[1] * f2[(base + id[1]) * Ci + c]
      + w[2] * f2[(base + id[2]) * Ci + c];
  } else {
    v = skip[r * Cs + (c - Ci)];
  }
  out[i] = v;
}

__global__ void k_gtranspose(const float* __restrict__ l3p, float* __restrict__ out,
                             int B, int S, int C)
{ // out[b][c][s] = l3p[b][s][c]  (flat == reshape(-1,512))
  size_t total = (size_t)B * S * C;
  size_t i = (size_t)blockIdx.x * blockDim.x + threadIdx.x;
  if (i >= total) return;
  int s = (int)(i % S); size_t t = i / S;
  int c = (int)(t % C); int b = (int)(t / C);
  out[i] = l3p[((size_t)b * S + s) * C + c];
}

// ---------------------------------------------------------------------------
// Host orchestration
// ---------------------------------------------------------------------------
static void run_mlp(const float* const* prm, int pbase, const int* dims, int nd,
                    const float* X, float* tA, float* tB, float* dest,
                    unsigned short* wp, unsigned short* xp,
                    int M, hipStream_t s)
{
  const int T = 256;
  const float* in = X;
  int nl = nd - 1;
  for (int lyr = 0; lyr < nl; ++lyr) {
    float* out = (lyr == nl - 1) ? dest : ((lyr % 2 == 0) ? tA : tB);
    int K = dims[lyr], N = dims[lyr + 1];
    int KT = CDIV(K, 32), NT = CDIV(N, 16), Kp = KT * 32;
    size_t wtot = (size_t)NT * KT * 512;
    k_pack_w<<<CDIV(wtot, T), T, 0, s>>>(prm[pbase + 3*lyr], wp, K, N, KT, NT);
    size_t xtot = (size_t)M * Kp;
    k_pack_x<<<CDIV(xtot, T), T, 0, s>>>(in, xp, (size_t)M, K, Kp);
    dim3 grid(CDIV(M, 64), CDIV(N, 64));
    k_gemm_wmma<<<grid, 128, 0, s>>>(xp, wp, prm[pbase + 3*lyr + 1],
                                     prm[pbase + 3*lyr + 2], out, M, N, Kp, KT, NT);
    in = out;
  }
}

static const int SA1_D[4] = {6, 32, 32, 64};
static const int SA2_D[4] = {64, 64, 64, 128};
static const int SA3_D[4] = {128, 128, 128, 256};
static const int SA4_D[3] = {256, 256, 512};
static const int FP4_D[3] = {768, 256, 256};
static const int FP3_D[3] = {384, 256, 256};
static const int FP2_D[3] = {320, 256, 128};
static const int FP1_D[5] = {137, 128, 128, 128, 128};

extern "C" void kernel_launch(void* const* d_in, const int* in_sizes, int n_in,
                              void* d_out, int out_size, void* d_ws, size_t ws_size,
                              hipStream_t stream)
{
  (void)in_sizes; (void)out_size; (void)ws_size;
  const int B = 4, N0 = 16384;
  const float* xyz    = (const float*)d_in[0];
  const float* points = (const float*)d_in[1];

  // params flattened in dict insertion order: sa1,sa2,sa3,sa4,fp4,fp3,fp2,fp1;
  // each layer contributes (W, g, b).
  const float* prm[66];
  for (int i = 0; i < 66; ++i) prm[i] = (i + 2 < n_in) ? (const float*)d_in[i + 2] : nullptr;
  const int P_SA1 = 0, P_SA2 = 9, P_SA3 = 18, P_SA4 = 27;
  const int P_FP4 = 36, P_FP3 = 42, P_FP2 = 48, P_FP1 = 54;

  // bump allocator over workspace
  char* base = (char*)d_ws;
  size_t off = 0;
  auto alloc = [&](size_t bytes) -> void* {
    void* p = base + off;
    off = (off + bytes + 255) & ~(size_t)255;
    return p;
  };

  float* f0    = (float*)alloc((size_t)B * N0 * 6 * 4);
  float* skip0 = (float*)alloc((size_t)B * N0 * 9 * 4);
  float* l1x   = (float*)alloc((size_t)B * 1024 * 3 * 4);
  float* l1f   = (float*)alloc((size_t)B * 1024 * 64 * 4);
  float* l2x   = (float*)alloc((size_t)B * 256 * 3 * 4);
  float* l2f   = (float*)alloc((size_t)B * 256 * 128 * 4);
  float* l3x   = (float*)alloc((size_t)B * 64 * 3 * 4);
  float* l3f   = (float*)alloc((size_t)B * 64 * 256 * 4);
  float* l4f   = (float*)alloc((size_t)B * 512 * 4);
  float* l3p   = (float*)alloc((size_t)B * 64 * 256 * 4);
  float* l2p   = (float*)alloc((size_t)B * 256 * 256 * 4);
  float* l1p   = (float*)alloc((size_t)B * 1024 * 128 * 4);
  int*   fi    = (int*)  alloc((size_t)B * 1024 * 4);
  int*   gi    = (int*)  alloc((size_t)B * 1024 * 32 * 4);   // max over levels
  int*   idx3  = (int*)  alloc((size_t)B * N0 * 3 * 4);
  float* w3    = (float*)alloc((size_t)B * N0 * 3 * 4);
  unsigned short* wp = (unsigned short*)alloc((size_t)262144 * 2);   // packed W
  unsigned short* xp = (unsigned short*)alloc((size_t)11000000 * 2); // packed X
  float* bufA  = (float*)alloc((size_t)9000000 * 4);
  float* bufB  = (float*)alloc((size_t)9000000 * 4);
  float* bufC  = (float*)alloc((size_t)9000000 * 4);

  const int T = 256;
  // f0 = transpose(points); skip0 = [xyz, f0]
  {
    size_t tot = (size_t)B * N0 * 6;
    k_transpose_points<<<CDIV(tot, T), T, 0, stream>>>(points, f0, B, 6, N0);
    size_t tot2 = (size_t)B * N0 * 9;
    k_concat2<<<CDIV(tot2, T), T, 0, stream>>>(xyz, 3, f0, 6, skip0, (size_t)B * N0);
  }

  // ---- SA1: N=16384 -> S=1024, r=0.1, ns=32, 6->32->32->64 ----
  {
    const int N = N0, S = 1024, ns = 32; const float r2 = 0.01f;
    k_fps<<<B, FPS_T, 0, stream>>>(xyz, fi, N, S);
    k_gather_xyz<<<CDIV(B * S, T), T, 0, stream>>>(xyz, fi, l1x, B, N, S);
    k_ball_query<<<CDIV(B * S, T), T, 0, stream>>>(l1x, xyz, gi, B, S, N, r2, ns);
    size_t tot = (size_t)B * S * ns * 6;
    k_group<<<CDIV(tot, T), T, 0, stream>>>(f0, gi, bufA, B, S, ns, N, 6);
    run_mlp(prm, P_SA1, SA1_D, 4, bufA, bufB, bufC, bufA, wp, xp, B * S * ns, stream);
    size_t tp = (size_t)B * S * 64;
    k_maxpool<<<CDIV(tp, T), T, 0, stream>>>(bufA, l1f, (size_t)B * S, ns, 64);
  }
  // ---- SA2: 1024 -> 256, r=0.2, ns=64, 64->64->64->128 ----
  {
    const int N = 1024, S = 256, ns = 64; const float r2 = 0.04f;
    k_fps<<<B, FPS_T, 0, stream>>>(l1x, fi, N, S);
    k_gather_xyz<<<CDIV(B * S, T), T, 0, stream>>>(l1x, fi, l2x, B, N, S);
    k_ball_query<<<CDIV(B * S, T), T, 0, stream>>>(l2x, l1x, gi, B, S, N, r2, ns);
    size_t tot = (size_t)B * S * ns * 64;
    k_group<<<CDIV(tot, T), T, 0, stream>>>(l1f, gi, bufA, B, S, ns, N, 64);
    run_mlp(prm, P_SA2, SA2_D, 4, bufA, bufB, bufC, bufA, wp, xp, B * S * ns, stream);
    size_t tp = (size_t)B * S * 128;
    k_maxpool<<<CDIV(tp, T), T, 0, stream>>>(bufA, l2f, (size_t)B * S, ns, 128);
  }
  // ---- SA3: 256 -> 64, r=0.4, ns=128, 128->128->128->256 ----
  {
    const int N = 256, S = 64, ns = 128; const float r2 = 0.16f;
    k_fps<<<B, FPS_T, 0, stream>>>(l2x, fi, N, S);
    k_gather_xyz<<<CDIV(B * S, T), T, 0, stream>>>(l2x, fi, l3x, B, N, S);
    k_ball_query<<<CDIV(B * S, T), T, 0, stream>>>(l3x, l2x, gi, B, S, N, r2, ns);
    size_t tot = (size_t)B * S * ns * 128;
    k_group<<<CDIV(tot, T), T, 0, stream>>>(l2f, gi, bufA, B, S, ns, N, 128);
    run_mlp(prm, P_SA3, SA3_D, 4, bufA, bufB, bufC, bufA, wp, xp, B * S * ns, stream);
    size_t tp = (size_t)B * S * 256;
    k_maxpool<<<CDIV(tp, T), T, 0, stream>>>(bufA, l3f, (size_t)B * S, ns, 256);
  }
  // ---- SA4 (global): mlp 256->256->512 on (B,64,.), max over 64 ----
  {
    run_mlp(prm, P_SA4, SA4_D, 3, l3f, bufB, bufC, bufA, wp, xp, B * 64, stream);
    size_t tp = (size_t)B * 512;
    k_maxpool<<<CDIV(tp, T), T, 0, stream>>>(bufA, l4f, (size_t)B, 64, 512);
  }
  // ---- FP4 (tile): [l4f bcast(512), l3f(256)] -> 768->256->256 -> l3p ----
  {
    size_t tot = (size_t)B * 64 * 768;
    k_tileconcat<<<CDIV(tot, T), T, 0, stream>>>(l4f, l3f, bufA, B, 64, 512, 256);
    run_mlp(prm, P_FP4, FP4_D, 3, bufA, bufB, bufC, l3p, wp, xp, B * 64, stream);
  }
  // ---- FP3: interp l3p to 256 pts, concat l2f -> 384->256->256 -> l2p ----
  {
    const int n1 = 256, n2 = 64, Ci = 256, Cs = 128;
    k_three_nn<<<CDIV(B * n1, T), T, 0, stream>>>(l2x, l3x, idx3, w3, B, n1, n2);
    size_t tot = (size_t)B * n1 * (Ci + Cs);
    k_interp_concat<<<CDIV(tot, T), T, 0, stream>>>(l3p, idx3, w3, l2f, bufA, B, n1, n2, Ci, Cs);
    run_mlp(prm, P_FP3, FP3_D, 3, bufA, bufB, bufC, l2p, wp, xp, B * n1, stream);
  }
  // ---- FP2: interp l2p to 1024 pts, concat l1f -> 320->256->128 -> l1p ----
  {
    const int n1 = 1024, n2 = 256, Ci = 256, Cs = 64;
    k_three_nn<<<CDIV(B * n1, T), T, 0, stream>>>(l1x, l2x, idx3, w3, B, n1, n2);
    size_t tot = (size_t)B * n1 * (Ci + Cs);
    k_interp_concat<<<CDIV(tot, T), T, 0, stream>>>(l2p, idx3, w3, l1f, bufA, B, n1, n2, Ci, Cs);
    run_mlp(prm, P_FP2, FP2_D, 3, bufA, bufB, bufC, l1p, wp, xp, B * n1, stream);
  }
  // ---- FP1: interp l1p to 16384 pts, concat skip0 -> 137->128x4 -> d_out ----
  {
    const int n1 = N0, n2 = 1024, Ci = 128, Cs = 9;
    k_three_nn<<<CDIV(B * n1, T), T, 0, stream>>>(xyz, l1x, idx3, w3, B, n1, n2);
    size_t tot = (size_t)B * n1 * (Ci + Cs);
    k_interp_concat<<<CDIV(tot, T), T, 0, stream>>>(l1p, idx3, w3, skip0, bufA, B, n1, n2, Ci, Cs);
    run_mlp(prm, P_FP1, FP1_D, 5, bufA, bufB, bufC, (float*)d_out, wp, xp, B * n1, stream);
  }
  // ---- global features: transpose(l3p,(0,2,1)) flat after point features ----
  {
    float* gout = (float*)d_out + (size_t)B * N0 * 128;
    size_t tot = (size_t)B * 64 * 256;
    k_gtranspose<<<CDIV(tot, T), T, 0, stream>>>(l3p, gout, B, 64, 256);
  }
}
